// SequentialDPClustering_5781025980420
// MI455X (gfx1250) — compile-verified, matrix operand order checked by reading, not run on hardware
//
#include <hip/hip_runtime.h>
#include <hip/hip_bf16.h>
#include <math.h>

typedef __attribute__((ext_vector_type(16))) _Float16 v16h;
typedef __attribute__((ext_vector_type(8)))  _Float16 v8h;
typedef __attribute__((ext_vector_type(8)))  float    v8f;

#define BB 8
#define DD 256
#define TT 2048
#define KNN 10
#define BETA 0.2f
#define THRESH 0.7f
#define MAXSPAN 4

// ---------------------------------------------------------------------------
// Kernel 0: row-normalize Xt (= x transposed) and store as fp16 [B, T, D].
// Block handles 32 consecutive t for one batch; 256 threads = (tt 0..31, dd 0..7).
// ---------------------------------------------------------------------------
__global__ void k_normalize(const float* __restrict__ x, _Float16* __restrict__ xn) {
    const int b     = blockIdx.x / (TT / 32);
    const int tBase = (blockIdx.x % (TT / 32)) * 32;
    const int tid = threadIdx.x;
    const int tt = tid & 31;       // t within tile
    const int dd = tid >> 5;       // 0..7 (d group)
    const int t = tBase + tt;

    __shared__ float psum[8][32];
    __shared__ float rinv_sh[32];
    __shared__ _Float16 tile[32][DD + 2];   // +2 halves pad to dodge bank conflicts

    // pass 1: sum of squares for row t (over all 256 d)
    float acc = 0.f;
    const float* xb = x + (size_t)b * DD * TT;
    #pragma unroll 8
    for (int j = 0; j < 32; ++j) {
        int d = dd * 32 + j;
        float v = xb[(size_t)d * TT + t];   // consecutive tt -> coalesced
        acc += v * v;
    }
    psum[dd][tt] = acc;
    __syncthreads();
    if (dd == 0) {
        float s = 0.f;
        #pragma unroll
        for (int g = 0; g < 8; ++g) s += psum[g][tt];
        float nrm = sqrtf(s);
        rinv_sh[tt] = 1.0f / fmaxf(nrm, 1e-12f);
    }
    __syncthreads();

    // pass 2: scale, convert to fp16, stage transposed in LDS
    float rinv = rinv_sh[tt];
    #pragma unroll 8
    for (int j = 0; j < 32; ++j) {
        int d = dd * 32 + j;
        float v = xb[(size_t)d * TT + t];
        tile[tt][d] = (_Float16)(v * rinv);
    }
    __syncthreads();

    // write xn[b, t, d] with contiguous d per row (coalesced)
    for (int row = 0; row < 32; ++row) {
        xn[((size_t)b * TT + tBase + row) * DD + tid] = tile[row][tid];
    }
}

// ---------------------------------------------------------------------------
// Kernel 1: sim = (Xn * Xn^T + 1) * 0.5 via v_wmma_f32_16x16x32_f16,
// stored as fp16 (67 MB -> L2-resident for the downstream row scans).
// One wave computes a 32x64 output strip: 2 A-frags x 4 B-frags = 8 acc,
// K-loop of 8 -> 64 WMMAs per wave, 12 b128 loads per 8 WMMAs.
// Fragment layout per ISA 7.12.2 (16-bit A 16x32): lane = (r = lane&15 row,
// hi = lane>>4), holds k chunks [k0+hi*8, +8) and [k0+16+hi*8, +8).
// B (32x16, B = Xn^T) has the mirrored layout -> identical per-lane loads.
// ---------------------------------------------------------------------------
__device__ inline v16h load_frag(const _Float16* __restrict__ row, int k0, int hi) {
    union { v16h v; v8h h[2]; } u;
    u.h[0] = *(const v8h*)(row + k0 + hi * 8);
    u.h[1] = *(const v8h*)(row + k0 + 16 + hi * 8);
    return u.v;
}

__global__ void k_simgemm(const _Float16* __restrict__ xn, _Float16* __restrict__ sim) {
    const int wid  = blockIdx.x * (blockDim.x >> 5) + (threadIdx.x >> 5);
    const int lane = threadIdx.x & 31;
    // wave-job decomposition: b (8) x bi (64 row blocks of 32) x jg (32 col groups of 64)
    const int b   = wid / (64 * 32);
    const int rem = wid % (64 * 32);
    const int bi  = rem / 32;
    const int jg  = rem % 32;

    const int r  = lane & 15;
    const int hi = lane >> 4;

    const _Float16* Xb = xn + (size_t)b * TT * DD;
    const _Float16* arow0 = Xb + (size_t)(bi * 32 + 0  + r) * DD;
    const _Float16* arow1 = Xb + (size_t)(bi * 32 + 16 + r) * DD;
    const _Float16* brow0 = Xb + (size_t)(jg * 64 + 0  + r) * DD;
    const _Float16* brow1 = Xb + (size_t)(jg * 64 + 16 + r) * DD;
    const _Float16* brow2 = Xb + (size_t)(jg * 64 + 32 + r) * DD;
    const _Float16* brow3 = Xb + (size_t)(jg * 64 + 48 + r) * DD;

    v8f c00 = {}, c01 = {}, c02 = {}, c03 = {};
    v8f c10 = {}, c11 = {}, c12 = {}, c13 = {};
    #pragma unroll
    for (int k0 = 0; k0 < DD; k0 += 32) {
        v16h a0 = load_frag(arow0, k0, hi);
        v16h a1 = load_frag(arow1, k0, hi);
        v16h b0 = load_frag(brow0, k0, hi);
        v16h b1 = load_frag(brow1, k0, hi);
        v16h b2 = load_frag(brow2, k0, hi);
        v16h b3 = load_frag(brow3, k0, hi);
        c00 = __builtin_amdgcn_wmma_f32_16x16x32_f16(false, a0, false, b0, (short)0, c00, false, false);
        c01 = __builtin_amdgcn_wmma_f32_16x16x32_f16(false, a0, false, b1, (short)0, c01, false, false);
        c02 = __builtin_amdgcn_wmma_f32_16x16x32_f16(false, a0, false, b2, (short)0, c02, false, false);
        c03 = __builtin_amdgcn_wmma_f32_16x16x32_f16(false, a0, false, b3, (short)0, c03, false, false);
        c10 = __builtin_amdgcn_wmma_f32_16x16x32_f16(false, a1, false, b0, (short)0, c10, false, false);
        c11 = __builtin_amdgcn_wmma_f32_16x16x32_f16(false, a1, false, b1, (short)0, c11, false, false);
        c12 = __builtin_amdgcn_wmma_f32_16x16x32_f16(false, a1, false, b2, (short)0, c12, false, false);
        c13 = __builtin_amdgcn_wmma_f32_16x16x32_f16(false, a1, false, b3, (short)0, c13, false, false);
    }

    // D layout: VGPR i -> row = rowBase + hi*8 + i, col = colBase + r
    _Float16* simb = sim + (size_t)b * TT * TT;
    #pragma unroll
    for (int i = 0; i < 8; ++i) {
        size_t row0 = (size_t)(bi * 32 + 0  + hi * 8 + i) * TT;
        size_t row1 = (size_t)(bi * 32 + 16 + hi * 8 + i) * TT;
        simb[row0 + jg * 64 + 0  + r] = (_Float16)((c00[i] + 1.f) * 0.5f);
        simb[row0 + jg * 64 + 16 + r] = (_Float16)((c01[i] + 1.f) * 0.5f);
        simb[row0 + jg * 64 + 32 + r] = (_Float16)((c02[i] + 1.f) * 0.5f);
        simb[row0 + jg * 64 + 48 + r] = (_Float16)((c03[i] + 1.f) * 0.5f);
        simb[row1 + jg * 64 + 0  + r] = (_Float16)((c10[i] + 1.f) * 0.5f);
        simb[row1 + jg * 64 + 16 + r] = (_Float16)((c11[i] + 1.f) * 0.5f);
        simb[row1 + jg * 64 + 32 + r] = (_Float16)((c12[i] + 1.f) * 0.5f);
        simb[row1 + jg * 64 + 48 + r] = (_Float16)((c13[i] + 1.f) * 0.5f);
    }
}

// ---------------------------------------------------------------------------
// Kernel 2: rho = exp(-mean(top-(K+1) of sim row, dropping the largest/self)).
// One wave per row; each lane scans 8-half vectors (b128-coalesced) keeping an
// insertion-sorted top-11 list; 11 rounds of wave-argmax merge pop the row top-11.
// ---------------------------------------------------------------------------
__global__ void k_rho(const _Float16* __restrict__ sim, float* __restrict__ rho) {
    const int row  = blockIdx.x * (blockDim.x >> 5) + (threadIdx.x >> 5); // 0..B*T-1
    const int lane = threadIdx.x & 31;
    const _Float16* srow = sim + (size_t)row * TT;

    float best[KNN + 1];
    #pragma unroll
    for (int j = 0; j <= KNN; ++j) best[j] = -1e30f;

    for (int i = 0; i < TT / 256; ++i) {           // 8 iterations, 8 halves/lane
        if (i + 1 < TT / 256)
            __builtin_prefetch(srow + (size_t)(i + 1) * 256 + lane * 8, 0, 1);
        v8h v8 = *(const v8h*)(srow + (size_t)i * 256 + lane * 8);
        #pragma unroll
        for (int e = 0; e < 8; ++e) {
            float v = (float)v8[e];
            if (v > best[KNN]) {
                best[KNN] = v;
                #pragma unroll
                for (int j = KNN; j > 0; --j) {
                    if (best[j] > best[j - 1]) {
                        float t = best[j - 1]; best[j - 1] = best[j]; best[j] = t;
                    }
                }
            }
        }
    }

    // merge across lanes: 11 pops of global max
    int ptr = 0;
    float sum = 0.f;
    for (int rnd = 0; rnd <= KNN; ++rnd) {
        float v = -1e30f;
        #pragma unroll
        for (int j = 0; j <= KNN; ++j) if (ptr == j) v = best[j];
        float wv = v; int wl = lane;
        #pragma unroll
        for (int off = 16; off > 0; off >>= 1) {
            float ov = __shfl_xor(wv, off, 32);
            int   ol = __shfl_xor(wl, off, 32);
            if (ov > wv || (ov == wv && ol < wl)) { wv = ov; wl = ol; }
        }
        if (wl == lane) ptr++;
        if (rnd > 0) sum += wv;   // rnd 0 = self (largest), dropped
    }
    if (lane == 0) rho[row] = __expf(-sum * (1.0f / KNN));
}

// ---------------------------------------------------------------------------
// Kernel 3: delta + s. One block per row, single vectorized pass (256 thr x v8h).
// delta_i = min_j{sim[i,j] : rho_j > rho_i} else max_j sim[i,j];  s = rho*delta.
// ---------------------------------------------------------------------------
__global__ void k_delta(const _Float16* __restrict__ sim, const float* __restrict__ rho,
                        float* __restrict__ s) {
    const int row = blockIdx.x;            // 0..B*T-1
    const int b = row / TT;
    const int tid = threadIdx.x;
    const _Float16* srow = sim + (size_t)row * TT;
    const float* rb = rho + (size_t)b * TT;
    const float ri = rho[row];

    float mnw = 1e30f, mxa = -1e30f;
    v8h v8 = *(const v8h*)(srow + tid * 8);
    #pragma unroll
    for (int e = 0; e < 8; ++e) {
        float v = (float)v8[e];
        if (rb[tid * 8 + e] > ri) mnw = fminf(mnw, v);
        mxa = fmaxf(mxa, v);
    }
    __shared__ float red_mn[256], red_mx[256];
    red_mn[tid] = mnw; red_mx[tid] = mxa;
    __syncthreads();
    for (int off = 128; off > 0; off >>= 1) {
        if (tid < off) {
            red_mn[tid] = fminf(red_mn[tid], red_mn[tid + off]);
            red_mx[tid] = fmaxf(red_mx[tid], red_mx[tid + off]);
        }
        __syncthreads();
    }
    if (tid == 0) {
        float delta = (red_mn[0] < 1e29f) ? red_mn[0] : red_mx[0];
        s[row] = ri * delta;
    }
}

// ---------------------------------------------------------------------------
// Kernel 4: sequential density-peak clustering, one block (256 thr) per batch.
// Produces rank-remapped ids [T], lens/starts sorted by cluster start.
// ---------------------------------------------------------------------------
__global__ void k_cluster(const _Float16* __restrict__ sim, const float* __restrict__ s,
                          int* __restrict__ ids_g, int* __restrict__ lens_g,
                          int* __restrict__ starts_g) {
    const int b = blockIdx.x;
    const int tid = threadIdx.x;
    const _Float16* simb = sim + (size_t)b * TT * TT;
    const float* sb = s + (size_t)b * TT;

    __shared__ float s_sh[TT];
    __shared__ int ids_sh[TT];
    __shared__ int starts_sh[TT];
    __shared__ int lens_sh[TT];
    __shared__ int rank_sh[TT];
    __shared__ unsigned char asg[TT];
    __shared__ float red_f[256];
    __shared__ int red_i[256];
    __shared__ int ctrl[2];   // [0]=remaining, [1]=cid

    for (int t = tid; t < TT; t += 256) {
        s_sh[t] = sb[t];
        ids_sh[t] = -1;
        starts_sh[t] = TT;
        lens_sh[t] = 0;
        asg[t] = 0;
    }
    if (tid == 0) { ctrl[0] = TT; ctrl[1] = 0; }
    __syncthreads();

    for (;;) {
        if (ctrl[0] <= 0) break;

        // masked argmax of s (first occurrence wins, as jnp.argmax)
        float bv = -1e30f; int bidx = 0;
        for (int t = tid; t < TT; t += 256) {
            float v = asg[t] ? -1e9f : s_sh[t];
            if (v > bv) { bv = v; bidx = t; }
        }
        red_f[tid] = bv; red_i[tid] = bidx;
        __syncthreads();
        for (int off = 128; off > 0; off >>= 1) {
            if (tid < off) {
                float ov = red_f[tid + off]; int oi = red_i[tid + off];
                if (ov > red_f[tid] || (ov == red_f[tid] && oi < red_i[tid])) {
                    red_f[tid] = ov; red_i[tid] = oi;
                }
            }
            __syncthreads();
        }

        if (tid == 0) {
            int seed = red_i[0];
            int cid = ctrl[1];
            asg[seed] = 1; ids_sh[seed] = cid;
            int remaining = ctrl[0] - 1;
            int size = 1, start = seed;
            // forward growth
            bool alive = true;
            for (int off = 1; off <= MAXSPAN; ++off) {
                int t = seed + off;
                int tc = t < TT - 1 ? t : TT - 1;
                float score = (float)simb[(size_t)seed * TT + tc] - BETA * s_sh[tc];
                bool ok = alive && (t < TT) && !asg[tc] && (size < MAXSPAN) && (score > THRESH);
                if (ok) { asg[tc] = 1; ids_sh[tc] = cid; size++; remaining--; }
                alive = ok;
            }
            // backward growth
            alive = true;
            for (int off = 1; off <= MAXSPAN; ++off) {
                int t = seed - off;
                int tc = t > 0 ? t : 0;
                float score = (float)simb[(size_t)seed * TT + tc] - BETA * s_sh[tc];
                bool ok = alive && (t >= 0) && !asg[tc] && (size < MAXSPAN) && (score > THRESH);
                if (ok) { asg[tc] = 1; ids_sh[tc] = cid; start = tc; size++; remaining--; }
                alive = ok;
            }
            lens_sh[cid] = size;
            starts_sh[cid] = start;
            ctrl[1] = cid + 1;
            ctrl[0] = remaining;
        }
        __syncthreads();
    }

    // rank clusters by start (stable: ties by original cid); unused (start=T) last
    for (int c = tid; c < TT; c += 256) {
        int sc = starts_sh[c];
        int cnt = 0;
        for (int c2 = 0; c2 < TT; ++c2) {
            int s2 = starts_sh[c2];
            cnt += (s2 < sc) || (s2 == sc && c2 < c);
        }
        rank_sh[c] = cnt;
        lens_g[(size_t)b * TT + cnt] = lens_sh[c];
        starts_g[(size_t)b * TT + cnt] = sc;
    }
    __syncthreads();
    for (int t = tid; t < TT; t += 256) {
        ids_g[(size_t)b * TT + t] = rank_sh[ids_sh[t]];
    }
}

// ---------------------------------------------------------------------------
// Kernel 5: clusters are contiguous spans -> mean-pool span [start, start+len)
// of x[b, :, t] into out[b, :, r]; lens written after the embeddings.
// One block per (b, cluster slot r); 256 threads cover D.
// ---------------------------------------------------------------------------
__global__ void k_pool(const float* __restrict__ x, const int* __restrict__ lens_g,
                       const int* __restrict__ starts_g, float* __restrict__ out) {
    const int b = blockIdx.x / TT;
    const int r = blockIdx.x % TT;
    const int d = threadIdx.x;

    const int len = lens_g[(size_t)b * TT + r];
    const int start = starts_g[(size_t)b * TT + r];

    float acc = 0.f;
    for (int k = 0; k < len; ++k) {
        int t = start + k;
        if (t < TT) acc += x[((size_t)b * DD + d) * TT + t];
    }
    float mean = acc / (float)(len > 0 ? len : 1);
    out[((size_t)b * DD + d) * TT + r] = mean;

    if (d == 0) {
        float* outLens = out + (size_t)BB * DD * TT;
        outLens[(size_t)b * TT + r] = (float)len;
    }
}

// ---------------------------------------------------------------------------
extern "C" void kernel_launch(void* const* d_in, const int* in_sizes, int n_in,
                              void* d_out, int out_size, void* d_ws, size_t ws_size,
                              hipStream_t stream) {
    const float* x = (const float*)d_in[0];
    float* out = (float*)d_out;

    // workspace carve-up
    size_t need = 0;
    _Float16* xn = (_Float16*)d_ws;                 need += (size_t)BB * TT * DD * sizeof(_Float16);
    _Float16* sim = (_Float16*)((char*)d_ws + need); need += (size_t)BB * TT * TT * sizeof(_Float16);
    float* rho = (float*)((char*)d_ws + need);       need += (size_t)BB * TT * sizeof(float);
    float* s   = (float*)((char*)d_ws + need);       need += (size_t)BB * TT * sizeof(float);
    int* ids   = (int*)((char*)d_ws + need);         need += (size_t)BB * TT * sizeof(int);
    int* lens  = (int*)((char*)d_ws + need);         need += (size_t)BB * TT * sizeof(int);
    int* starts = (int*)((char*)d_ws + need);        need += (size_t)BB * TT * sizeof(int);
    if (ws_size < need) return;

    // 0) normalize + fp16
    k_normalize<<<BB * (TT / 32), 256, 0, stream>>>(x, xn);

    // 1) WMMA sim GEMM: wave-jobs = B*64*32 = 16384 -> 2048 blocks of 8 waves
    k_simgemm<<<(BB * 64 * 32) / 8, 256, 0, stream>>>(xn, sim);

    // 2) rho: one wave per row, 8 waves/block
    k_rho<<<(BB * TT) / 8, 256, 0, stream>>>(sim, rho);

    // 3) delta + s: one block per row
    k_delta<<<BB * TT, 256, 0, stream>>>(sim, rho, s);

    // 4) sequential clustering: one block per batch
    k_cluster<<<BB, 256, 0, stream>>>(sim, s, ids, lens, starts);

    // 5) span mean-pool + lens output
    k_pool<<<BB * TT, 256, 0, stream>>>(x, lens, starts, out);
}
